// LinkPredictor_53060025975161
// MI455X (gfx1250) — matrix operand order repaired and checked by reading, not compiled, not run on hardware
//
#include <hip/hip_runtime.h>
#include <math.h>

// ---------------------------------------------------------------------------
// CDNA5 (gfx1250) wave32 WMMA / TDM types
// ---------------------------------------------------------------------------
typedef __attribute__((ext_vector_type(16))) __bf16        v16bf;
typedef __attribute__((ext_vector_type(8)))  float         v8f;
typedef __attribute__((ext_vector_type(4)))  unsigned int  v4u;
typedef __attribute__((ext_vector_type(8)))  int           v8i;
typedef __attribute__((ext_vector_type(4)))  int           v4i;

// ---------------------------------------------------------------------------
// TDM: DMA a 2D f32 tile (tile_w x tile_h, row stride stride_w elems) from
// global memory into LDS at byte offset lds_addr. D# layout per CDNA5 ISA
// 8.3/8.4: group0 = {count, lds_addr, global_addr(57b), type=2},
// group1 = {data_size=4B, tensor dims, tile dims, dim0 stride}, groups 2/3 = 0.
// This toolchain exposes the 6-arg builtin:
//   (uint32x4 g0, int32x8 g1, int32x4, int32x4, int32x8, i32 cpol)
// Issued by one wave; EXEC is ignored by TENSOR_LOAD_TO_LDS.
// ---------------------------------------------------------------------------
__device__ __forceinline__ void tdm_load_2d_f32(unsigned int lds_addr,
                                                const float* gptr,
                                                int tensor_w, int tensor_h,
                                                int stride_w,
                                                int tile_w, int tile_h) {
  const unsigned long long ga = (unsigned long long)(const void*)gptr;
  v4u g0;
  g0[0] = 1u;                                            // count=1, user mode
  g0[1] = lds_addr;                                      // LDS byte address
  g0[2] = (unsigned int)(ga & 0xffffffffull);            // global_addr[31:0]
  g0[3] = (unsigned int)((ga >> 32) & 0x1ffffffull)      // global_addr[56:32]
          | (2u << 30);                                  // type=2 ("image")
  v8i g1;
  g1[0] = 0x20000;                                       // data_size=2 -> 4B
  g1[1] = (tensor_w & 0xffff) << 16;                     // tensor_dim0 lo16
  g1[2] = ((tensor_w >> 16) & 0xffff)                    // tensor_dim0 hi16
          | ((tensor_h & 0xffff) << 16);                 // tensor_dim1 lo16
  g1[3] = ((tensor_h >> 16) & 0xffff)                    // tensor_dim1 hi16
          | ((tile_w & 0xffff) << 16);                   // tile_dim0
  g1[4] = (tile_h & 0xffff);                             // tile_dim1 (dim2=0)
  g1[5] = stride_w;                                      // dim0_stride lo32
  g1[6] = 0;                                             // stride hi / dim1_stride
  g1[7] = 0;
  v4i g2 = {0, 0, 0, 0};                                 // 2D: groups 2/3 unused
  v4i g3 = {0, 0, 0, 0};
  v8i g4 = {0, 0, 0, 0, 0, 0, 0, 0};                     // extra group (unused)
  __builtin_amdgcn_tensor_load_to_lds(g0, g1, g2, g3, g4, 0);
}

// ---------------------------------------------------------------------------
// WMMA operand loaders, per CDNA5 ISA wave32 layouts.
// A (16x32 bf16, MxK): lanes 0-15 -> M=lane, lanes 16-31 -> M=lane-16;
//   VGPR v holds K = (v/4)*16 + (lane>=16 ? 8 : 0) + (v%4)*2 .. +1
// B (32x16 bf16, KxN): lane holds column N=lane&15; VGPR v packs K as for A.
// C/D (16x16 f32): n = lane&15, m = vgpr + 8*(lane>=16).
// ---------------------------------------------------------------------------
__device__ __forceinline__ v16bf load_A_bf16(const float* __restrict__ A,
                                             int lda, int m0, int k0, int lane) {
  const int half = lane >> 4;
  const int ml   = lane & 15;
  const float* row = A + (size_t)(m0 + ml) * (size_t)lda;
  v16bf a;
#pragma unroll
  for (int v = 0; v < 8; ++v) {
    const int k = k0 + ((v & 4) << 2) + (half << 3) + ((v & 3) << 1);
    a[2 * v]     = (__bf16)row[k];
    a[2 * v + 1] = (__bf16)row[k + 1];
  }
  return a;
}

__device__ __forceinline__ v16bf load_B_lds_bf16(const float* brow, int ld, int half) {
  v16bf b;
#pragma unroll
  for (int v = 0; v < 8; ++v) {
    const int k = ((v & 4) << 2) + (half << 3) + ((v & 3) << 1);
    b[2 * v]     = (__bf16)brow[k * ld];
    b[2 * v + 1] = (__bf16)brow[(k + 1) * ld];
  }
  return b;
}

// ---------------------------------------------------------------------------
// Dense GEMM: C[M,N] = A[M,K] @ B[K,N], f32 in/out, bf16 WMMA compute.
// Grid: x = M/64 (4 waves/block, one 16-row tile each), y = N/16.
// The block's Kx16 B slice is TDM-staged into LDS once and reused by 4 waves.
// ---------------------------------------------------------------------------
__global__ __launch_bounds__(128) void gemm_bf16_wmma(
    const float* __restrict__ A, const float* __restrict__ B,
    float* __restrict__ C, int M, int N, int K) {
  extern __shared__ float ldsB[];                 // K x 16 f32 slice
  const int wid  = threadIdx.x >> 5;
  const int lane = threadIdx.x & 31;
  const int n0   = blockIdx.y << 4;
  const int m0   = (blockIdx.x * 4 + wid) << 4;

  if (wid == 0) {                                 // wave-uniform: one DMA/block
    tdm_load_2d_f32((unsigned int)(size_t)(void*)ldsB, B + n0,
                    /*tensor_w=*/N, /*tensor_h=*/K, /*stride_w=*/N,
                    /*tile_w=*/16, /*tile_h=*/K);
    __builtin_amdgcn_s_wait_tensorcnt(0);
  }
  __syncthreads();

  if (m0 < M) {                                   // wave-uniform guard
    v8f c = {};
    for (int k0 = 0; k0 < K; k0 += 32) {
      v16bf a = load_A_bf16(A, K, m0, k0, lane);
      v16bf b = load_B_lds_bf16(ldsB + (size_t)k0 * 16 + (lane & 15), 16, lane >> 4);
      c = __builtin_amdgcn_wmma_f32_16x16x32_bf16(
          false, a, false, b, (short)0, c, false, false);
    }
    const int half = lane >> 4, nl = lane & 15;
#pragma unroll
    for (int v = 0; v < 8; ++v) {
      C[(size_t)(m0 + v + (half << 3)) * (size_t)N + (size_t)(n0 + nl)] = c[v];
    }
  }
}

// ---------------------------------------------------------------------------
// Degree / normalization helpers
// ---------------------------------------------------------------------------
__global__ void fill_one_kernel(float* __restrict__ p, int n) {
  int t = blockIdx.x * blockDim.x + threadIdx.x;
  if (t < n) p[t] = 1.0f;  // self-loop: deg starts at 1
}

__global__ void deg_count_kernel(const int* __restrict__ dst,
                                 float* __restrict__ deg, int E) {
  int t = blockIdx.x * blockDim.x + threadIdx.x;
  if (t < E) atomicAdd(&deg[dst[t]], 1.0f);
}

__global__ void rsqrt_inplace_kernel(float* __restrict__ p, int n) {
  int t = blockIdx.x * blockDim.x + threadIdx.x;
  if (t < n) p[t] = rsqrtf(p[t]);
}

// ---------------------------------------------------------------------------
// GCN aggregation: agg = selfloop + scatter(messages). L2-resident atomics.
// ---------------------------------------------------------------------------
__global__ void selfloop_init_kernel(const float* __restrict__ hlin,
                                     const float* __restrict__ dis,
                                     float* __restrict__ agg, int N, int D) {
  int t = blockIdx.x * blockDim.x + threadIdx.x;
  const int chunks = D >> 2;
  if (t >= N * chunks) return;
  const int i = t / chunks;
  const int d = (t - i * chunks) << 2;
  float s = dis[i];
  s = s * s;
  const float4 h = *(const float4*)(hlin + (size_t)i * D + d);
  float4 o;
  o.x = h.x * s; o.y = h.y * s; o.z = h.z * s; o.w = h.w * s;
  *(float4*)(agg + (size_t)i * D + d) = o;
}

__global__ void scatter_edges_kernel(const float* __restrict__ hlin,
                                     const int* __restrict__ src,
                                     const int* __restrict__ dst,
                                     const float* __restrict__ dis,
                                     float* __restrict__ agg, int E, int D) {
  const long long t = (long long)blockIdx.x * blockDim.x + threadIdx.x;
  const int chunks = D >> 2;
  const long long total = (long long)E * chunks;
  if (t >= total) return;
  const int e = (int)(t / chunks);
  const int d = (int)(t - (long long)e * chunks) << 2;
  const int s  = src[e];
  const int dd = dst[e];
  const float coef = dis[s] * dis[dd];
  const float4 h = *(const float4*)(hlin + (size_t)s * D + d);
  float* base = agg + (size_t)dd * D + d;
  atomicAdd(base + 0, h.x * coef);
  atomicAdd(base + 1, h.y * coef);
  atomicAdd(base + 2, h.z * coef);
  atomicAdd(base + 3, h.w * coef);
}

__global__ void bias_act_kernel(float* __restrict__ a, const float* __restrict__ b,
                                int N, int D, int do_relu) {
  int t = blockIdx.x * blockDim.x + threadIdx.x;
  if (t >= N * D) return;
  const int d = t % D;
  float v = a[t] + b[d];
  if (do_relu) v = fmaxf(v, 0.0f);
  a[t] = v;
}

// ---------------------------------------------------------------------------
// Fused eval-edge MLP: TDM-stage Wf1 (128x64 f32, 32 KB) into LDS once per
// 8-wave block; each wave gathers z[es]||z[ed] (K=128), does 16 WMMAs,
// bias+ReLU in registers, 64->1 dot via 16-lane shfl butterflies, sigmoid.
// Hidden activations never touch memory.
// ---------------------------------------------------------------------------
__global__ __launch_bounds__(256) void edge_mlp_wmma_kernel(
    const float* __restrict__ z, const int* __restrict__ es,
    const int* __restrict__ ed, const float* __restrict__ Wf1,
    const float* __restrict__ bf1, const float* __restrict__ Wf2,
    const float* __restrict__ bf2, float* __restrict__ out, int E) {
  __shared__ float ldsW[128 * 64];                // full Wf1, f32
  const int wid  = threadIdx.x >> 5;
  const int lane = threadIdx.x & 31;
  const int m0   = (blockIdx.x * 8 + wid) << 4;

  if (wid == 0) {
    tdm_load_2d_f32((unsigned int)(size_t)(void*)ldsW, Wf1,
                    /*tensor_w=*/64, /*tensor_h=*/128, /*stride_w=*/64,
                    /*tile_w=*/64, /*tile_h=*/128);
    __builtin_amdgcn_s_wait_tensorcnt(0);
  }
  __syncthreads();

  if (m0 < E) {                                   // wave-uniform
    const int half = lane >> 4;
    const int ml   = lane & 15;
    const int m    = m0 + ml;
    const float* rowS = z + (size_t)es[m] * 64;   // first 64 feats of pair
    const float* rowD = z + (size_t)ed[m] * 64;   // last 64 feats of pair

    float lacc[8] = {0.f, 0.f, 0.f, 0.f, 0.f, 0.f, 0.f, 0.f};

#pragma unroll
    for (int nt = 0; nt < 4; ++nt) {              // N = 64 -> 4 tiles
      const int n0 = nt << 4;
      v8f c = {};
#pragma unroll
      for (int kt = 0; kt < 4; ++kt) {            // K = 128 -> 4 steps
        const int k0 = kt << 5;
        v16bf a;
#pragma unroll
        for (int v = 0; v < 8; ++v) {
          const int k = k0 + ((v & 4) << 2) + (half << 3) + ((v & 3) << 1);
          const float* r = (k < 64) ? rowS : rowD;  // concat [z[es], z[ed]]
          const int kk = k & 63;                    // pair stays in one half
          a[2 * v]     = (__bf16)r[kk];
          a[2 * v + 1] = (__bf16)r[kk + 1];
        }
        v16bf b = load_B_lds_bf16(ldsW + (size_t)k0 * 64 + n0 + ml, 64, half);
        c = __builtin_amdgcn_wmma_f32_16x16x32_bf16(
            false, a, false, b, (short)0, c, false, false);
      }
      const float bias = bf1[n0 + ml];
      const float w2   = Wf2[n0 + ml];
#pragma unroll
      for (int v = 0; v < 8; ++v) {
        float hv = c[v] + bias;
        hv = fmaxf(hv, 0.0f);
        lacc[v] += hv * w2;
      }
    }

    const float b2 = bf2[0];
#pragma unroll
    for (int v = 0; v < 8; ++v) {
      float s = lacc[v];
      s += __shfl_xor(s, 1, 16);
      s += __shfl_xor(s, 2, 16);
      s += __shfl_xor(s, 4, 16);
      s += __shfl_xor(s, 8, 16);
      if (ml == 0) {
        const int mo = m0 + (half << 3) + v;      // half0: m0+v, half1: m0+8+v
        out[mo] = 1.0f / (1.0f + __expf(-(s + b2)));
      }
    }
  }
}

// ---------------------------------------------------------------------------
// Host-side orchestration
// ---------------------------------------------------------------------------
static inline size_t align256(size_t x) { return (x + 255) & ~(size_t)255; }
static inline unsigned cdiv(long long a, long long b) { return (unsigned)((a + b - 1) / b); }

extern "C" void kernel_launch(void* const* d_in, const int* in_sizes, int n_in,
                              void* d_out, int out_size, void* d_ws, size_t ws_size,
                              hipStream_t stream) {
  const float* x   = (const float*)d_in[0];
  const int*   ei  = (const int*)d_in[1];   // [2,E] int32 (JAX default x64 off)
  const int*   eei = (const int*)d_in[2];   // [2,E_eval]
  const float* W1  = (const float*)d_in[3];
  const float* b1  = (const float*)d_in[4];
  const float* W2  = (const float*)d_in[5];
  const float* b2  = (const float*)d_in[6];
  const float* Wf1 = (const float*)d_in[7];
  const float* bf1 = (const float*)d_in[8];
  const float* Wf2 = (const float*)d_in[9];
  const float* bf2 = (const float*)d_in[10];
  float* out = (float*)d_out;

  const int IN = 256, H = 128, L = 64;
  const int N  = in_sizes[0] / IN;          // 50000
  const int E  = in_sizes[1] / 2;           // 1,600,000
  const int Ee = in_sizes[2] / 2;           // 1,000,000
  const int* src = ei;
  const int* dst = ei + E;
  const int* es  = eei;
  const int* ed  = eei + Ee;

  // Workspace layout (with reuse):  deg[N] | buf1[N*H] | buf2[N*H]
  char* ws = (char*)d_ws;
  float* deg   = (float*)ws;
  float* hlin1 = (float*)(ws + align256((size_t)N * sizeof(float)));
  float* agg1  = hlin1 + (size_t)N * H;
  float* hlin2 = hlin1;  // hlin1 dead after layer-1 aggregation
  float* agg2  = agg1;   // agg1 (=h) dead after GEMM2

  const int TB = 256;

  // --- degree with self loops, deg_isqrt (in place) ---
  fill_one_kernel<<<cdiv(N, TB), TB, 0, stream>>>(deg, N);
  deg_count_kernel<<<cdiv(E, TB), TB, 0, stream>>>(dst, deg, E);
  rsqrt_inplace_kernel<<<cdiv(N, TB), TB, 0, stream>>>(deg, N);

  // --- layer 1: hlin1 = x @ W1 ; agg ; relu(+b1) ---
  {
    dim3 grid(cdiv(N / 16, 4), H / 16);
    gemm_bf16_wmma<<<grid, 128, (size_t)IN * 16 * sizeof(float), stream>>>(
        x, W1, hlin1, N, H, IN);
  }
  selfloop_init_kernel<<<cdiv((long long)N * (H / 4), TB), TB, 0, stream>>>(
      hlin1, deg, agg1, N, H);
  scatter_edges_kernel<<<cdiv((long long)E * (H / 4), TB), TB, 0, stream>>>(
      hlin1, src, dst, deg, agg1, E, H);
  bias_act_kernel<<<cdiv((long long)N * H, TB), TB, 0, stream>>>(agg1, b1, N, H, 1);

  // --- layer 2: hlin2 = h @ W2 ; agg ; +b2 -> z ---
  {
    dim3 grid(cdiv(N / 16, 4), L / 16);
    gemm_bf16_wmma<<<grid, 128, (size_t)H * 16 * sizeof(float), stream>>>(
        agg1, W2, hlin2, N, L, H);
  }
  selfloop_init_kernel<<<cdiv((long long)N * (L / 4), TB), TB, 0, stream>>>(
      hlin2, deg, agg2, N, L);
  scatter_edges_kernel<<<cdiv((long long)E * (L / 4), TB), TB, 0, stream>>>(
      hlin2, src, dst, deg, agg2, E, L);
  bias_act_kernel<<<cdiv((long long)N * L, TB), TB, 0, stream>>>(agg2, b2, N, L, 0);

  // --- fused eval-edge MLP + sigmoid ---
  {
    long long waves = (Ee + 15) / 16;
    edge_mlp_wmma_kernel<<<cdiv(waves, 8), 256, 0, stream>>>(
        agg2, es, ed, Wf1, bf1, Wf2, bf2, out, Ee);
  }
}
